// PerformerEncoder_85418309582958
// MI455X (gfx1250) — matrix operand order repaired and checked by reading, not compiled, not run on hardware
//
#include <hip/hip_runtime.h>

#define B_  4
#define N_  2048
#define D_  512
#define L_  6
#define H_  8
#define DH_ 64
#define M_  266
#define MP_ 272    // M padded to multiple of 8 halfs (16B rows)
#define FF_ 2048

typedef __attribute__((ext_vector_type(16))) _Float16 v16h;
typedef __attribute__((ext_vector_type(8)))  _Float16 v8h;
typedef __attribute__((ext_vector_type(8)))  float    v8f;

// ---------------------------------------------------------------------------
// wave32 reduction helpers
// ---------------------------------------------------------------------------
__device__ inline float wave_sum(float v) {
#pragma unroll
  for (int o = 16; o; o >>= 1) v += __shfl_xor(v, o, 32);
  return v;
}
__device__ inline float wave_max(float v) {
#pragma unroll
  for (int o = 16; o; o >>= 1) v = fmaxf(v, __shfl_xor(v, o, 32));
  return v;
}

// ---------------------------------------------------------------------------
// f16 strided batched GEMM:  C[b] = epi( A[b] @ B[b] )
// A: f16, element (m,k) at A[b*abs_ + m*ars + k]      (unit k stride)
// B: f16, element (k,n) at B[b*bbs  + k*brs + n]      (unit n stride)
// Outputs (each optional): C   fp32 row-major [Md x Nd] batch stride cbs
//                          C16 f16  row stride c16rs, batch stride c16bs
// epilogue: +bias[col], exact GELU, *row_scale[row], +residual
// 128x128 tile, 8 waves, each wave 2x4 wmma tiles, K-step 32.
// Interior blocks take a select-free staging fast path; edge blocks use the
// branchless clamped path. All f16 buffers are over-allocated (padded) so the
// clamped loads stay in-bounds.
// ---------------------------------------------------------------------------
#define TM 128
#define TN 128
#define TK 32
#define LPAD 40   // halfs per LDS row (80B, keeps 16B alignment of chunks)

__global__ void __launch_bounds__(256) gemm_wmma_k(
    const _Float16* __restrict__ A, const _Float16* __restrict__ Bm,
    float* __restrict__ C, _Float16* __restrict__ C16,
    int Md, int Nd, int Kd,
    long ars, long abs_, long brs, long bbs,
    long cbs, long c16rs, long c16bs,
    const float* __restrict__ bias,
    const float* __restrict__ row_scale, long rsbs,
    const float* __restrict__ residual, long resbs,
    int gelu_flag)
{
  __shared__ __align__(16) _Float16 As[TM * LPAD];
  __shared__ __align__(16) _Float16 Bs[TN * LPAD];   // transposed: Bs[col][k]

  const int tid   = threadIdx.x;
  const int lane  = tid & 31;
  const int wid   = tid >> 5;
  const int batch = blockIdx.z;
  const int m0 = blockIdx.y * TM;
  const int n0 = blockIdx.x * TN;

  A   += (long)batch * abs_;
  Bm  += (long)batch * bbs;
  if (C)   C   += (long)batch * cbs;
  if (C16) C16 += (long)batch * c16bs;

  const int wm0 = (wid >> 1) * 32;   // 4 wave-rows * 32
  const int wn0 = (wid & 1) * 64;    // 2 wave-cols * 64

  // per-block uniform: whole tile in-bounds and K a multiple of TK
  const bool interior = (m0 + TM <= Md) && (n0 + TN <= Nd) && ((Kd & (TK - 1)) == 0);

  v8f zero = {};
  v8f acc[2][4];
#pragma unroll
  for (int i = 0; i < 2; ++i)
#pragma unroll
    for (int j = 0; j < 4; ++j) acc[i][j] = zero;

  // per-thread staging coordinates
  const int ar0 = tid >> 2;            // A row (first chunk)
  const int akc = (tid & 3) << 3;      // A k-chunk offset
  const int bk0 = tid >> 4;            // B k row (first chunk)
  const int bnc = (tid & 15) << 3;     // B n-chunk offset

  const int kTiles = (Kd + TK - 1) / TK;
  for (int kt = 0; kt < kTiles; ++kt) {
    const int k0 = kt * TK;

    if (interior) {
      // ---- fast path: pure vector copies, no selects ----
#pragma unroll
      for (int it = 0; it < 2; ++it) {
        int r  = ar0 + it * 64;
        const _Float16* gp = A + (long)(m0 + r) * ars + (k0 + akc);
        v8h a = *(const v8h*)gp;
        __builtin_prefetch((const void*)(gp + TK), 0, 1);   // next K tile
        *(v8h*)(&As[r * LPAD + akc]) = a;
      }
#pragma unroll
      for (int it = 0; it < 2; ++it) {
        int kk = bk0 + it * 16;
        const _Float16* gp = Bm + (long)(k0 + kk) * brs + (n0 + bnc);
        v8h b = *(const v8h*)gp;
        __builtin_prefetch((const void*)(gp + (long)TK * brs), 0, 1);
#pragma unroll
        for (int u = 0; u < 8; ++u)
          Bs[(bnc + u) * LPAD + kk] = b[u];
      }
    } else {
      // ---- generic branchless path: clamped addresses + selects ----
#pragma unroll
      for (int it = 0; it < 2; ++it) {
        int r   = ar0 + it * 64;
        int gm  = m0 + r;
        int gk0 = k0 + akc;
        int gmc = (gm < Md) ? gm : (Md - 1);
        v8h a = *(const v8h*)(A + (long)gmc * ars + gk0);
        bool rok = (gm < Md);
#pragma unroll
        for (int u = 0; u < 8; ++u)
          a[u] = (rok && (gk0 + u) < Kd) ? a[u] : (_Float16)0.f;
        *(v8h*)(&As[r * LPAD + akc]) = a;
      }
#pragma unroll
      for (int it = 0; it < 2; ++it) {
        int kk  = bk0 + it * 16;
        int gk  = k0 + kk;
        int gn0 = n0 + bnc;
        int gkc = (gk < Kd) ? gk : (Kd - 1);
        v8h b = *(const v8h*)(Bm + (long)gkc * brs + gn0);
        bool kok = (gk < Kd);
#pragma unroll
        for (int u = 0; u < 8; ++u)
          b[u] = (kok && (gn0 + u) < Nd) ? b[u] : (_Float16)0.f;
#pragma unroll
        for (int u = 0; u < 8; ++u)
          Bs[(bnc + u) * LPAD + kk] = b[u];
      }
    }
    __syncthreads();

    // A fragments: lanes 0-15 -> K{0..7,16..23}; lanes 16-31 -> K{8..15,24..31}
    const int fr = lane & 15;
    const int ka = (lane < 16) ? 0 : 8;
    v16h afrag[2], bfrag[4];
#pragma unroll
    for (int i = 0; i < 2; ++i) {
      const _Float16* p = &As[(wm0 + i * 16 + fr) * LPAD + ka];
      v8h lo = *(const v8h*)p;
      v8h hi = *(const v8h*)(p + 16);
      afrag[i] = __builtin_shufflevector(lo, hi, 0,1,2,3,4,5,6,7,8,9,10,11,12,13,14,15);
    }
    // B fragments: lanes 0-15 -> K{0..7,16..23}; lanes 16-31 -> K{8..15,24..31}
    const int kb = (lane < 16) ? 0 : 8;
#pragma unroll
    for (int j = 0; j < 4; ++j) {
      const _Float16* p = &Bs[(wn0 + j * 16 + fr) * LPAD + kb];
      v8h lo = *(const v8h*)p;
      v8h hi = *(const v8h*)(p + 16);
      bfrag[j] = __builtin_shufflevector(lo, hi, 0,1,2,3,4,5,6,7,8,9,10,11,12,13,14,15);
    }
#pragma unroll
    for (int i = 0; i < 2; ++i)
#pragma unroll
      for (int j = 0; j < 4; ++j)
        acc[i][j] = __builtin_amdgcn_wmma_f32_16x16x32_f16(
            false, afrag[i], false, bfrag[j], (short)0, acc[i][j], false, false);
    __syncthreads();
  }

  // epilogue + store: lanes 0-15 -> M=e, lanes 16-31 -> M=8+e; N=lane&15
  const int cn  = lane & 15;
  const int cmb = (lane >> 4) * 8;
#pragma unroll
  for (int i = 0; i < 2; ++i) {
#pragma unroll
    for (int j = 0; j < 4; ++j) {
#pragma unroll
      for (int e = 0; e < 8; ++e) {
        int row = m0 + wm0 + i * 16 + cmb + e;
        int col = n0 + wn0 + j * 16 + cn;
        if (row < Md && col < Nd) {
          float val = acc[i][j][e];
          if (bias)      val += bias[col];
          if (gelu_flag) val = 0.5f * val * (1.f + erff(val * 0.70710678118654752f));
          if (row_scale) val *= row_scale[(long)batch * rsbs + row];
          if (residual)  val += residual[(long)batch * resbs + (long)row * Nd + col];
          if (C)   C[(long)row * Nd + col] = val;
          if (C16) C16[(long)row * c16rs + col] = (_Float16)val;
        }
      }
    }
  }
}

// ---------------------------------------------------------------------------
// Elementwise / reduction kernels
// ---------------------------------------------------------------------------
__global__ void __launch_bounds__(256) posadd_k(const float* __restrict__ xin,
                                                float* __restrict__ xw) {
  long idx = (long)blockIdx.x * 256 + threadIdx.x;
  if (idx >= (long)B_ * N_ * D_) return;
  int d = (int)(idx % D_);
  int n = (int)((idx / D_) % N_);
  int j = (d < D_ / 2) ? d : d - D_ / 2;
  float invf = expf(-logf(10000.f) * (2.f * j) / (float)D_);
  float s = (float)n * invf;
  float pe = (d < D_ / 2) ? sinf(s) : cosf(s);
  xw[idx] = xin[idx] + pe;
}

// fp32 -> f16 copy (weights)
__global__ void __launch_bounds__(256) cvt_k(const float* __restrict__ in,
                                             _Float16* __restrict__ out, long n) {
  long idx = (long)blockIdx.x * 256 + threadIdx.x;
  if (idx < n) out[idx] = (_Float16)in[idx];
}

// proj (M x DH) -> projT16 (DH x MP_) with transpose + pad
__global__ void __launch_bounds__(256) projT_k(const float* __restrict__ pj,
                                               _Float16* __restrict__ out) {
  int idx = blockIdx.x * 256 + threadIdx.x;
  if (idx >= DH_ * M_) return;
  int m = idx % M_;
  int kd = idx / M_;
  out[(long)kd * MP_ + m] = (_Float16)pj[(long)m * DH_ + kd];
}

// LayerNorm; optional fp32 and f16 outputs. One wave per row.
__global__ void __launch_bounds__(256) layernorm_k(
    const float* __restrict__ x, const float* __restrict__ g,
    const float* __restrict__ b, float* __restrict__ out,
    _Float16* __restrict__ out16, int rows, int dim) {
  int row = blockIdx.x * 8 + (threadIdx.x >> 5);
  int lane = threadIdx.x & 31;
  if (row >= rows) return;
  const float* xr = x + (long)row * dim;
  float s = 0.f;
  for (int i = lane; i < dim; i += 32) s += xr[i];
  float mu = wave_sum(s) / dim;
  float v = 0.f;
  for (int i = lane; i < dim; i += 32) { float d0 = xr[i] - mu; v += d0 * d0; }
  float inv = rsqrtf(wave_sum(v) / dim + 1e-5f);
  for (int i = lane; i < dim; i += 32) {
    float val = (xr[i] - mu) * inv * g[i] + b[i];
    if (out)   out[(long)row * dim + i] = val;
    if (out16) out16[(long)row * dim + i] = (_Float16)val;
  }
}

// rotary + (b,n,h,d) -> (b,h,n,d) transpose, fp32 in -> f16 out, output
// pre-scaled by anorm = DH^-0.25 (folded Performer normalizer)
__global__ void __launch_bounds__(256) rotary_tr_k(const float* __restrict__ in,
                                                   _Float16* __restrict__ out) {
  long idx = (long)blockIdx.x * 256 + threadIdx.x;
  const long total = (long)B_ * N_ * H_ * (DH_ / 2);
  if (idx >= total) return;
  const float anorm = 0.35355339059327373f;
  int p = (int)(idx % (DH_ / 2));
  int h = (int)((idx / (DH_ / 2)) % H_);
  int n = (int)((idx / ((DH_ / 2) * H_)) % N_);
  int b = (int)(idx / ((long)(DH_ / 2) * H_ * N_));
  const float* src = in + (((long)(b * N_ + n) * H_ + h) * DH_) + 2 * p;
  float q0 = src[0], q1 = src[1];
  float invf = expf(-logf(10000.f) * (2.f * p) / (float)DH_);
  float s = sinf((float)n * invf), c = cosf((float)n * invf);
  _Float16* dst = out + (((long)(b * H_ + h) * N_ + n) * DH_) + 2 * p;
  dst[0] = (_Float16)((q0 * c - q1 * s) * anorm);
  dst[1] = (_Float16)((q1 * c + q0 * s) * anorm);
}

__global__ void __launch_bounds__(256) bnhd2bhnd_k(const float* __restrict__ in,
                                                   _Float16* __restrict__ out) {
  long idx = (long)blockIdx.x * 256 + threadIdx.x;
  if (idx >= (long)B_ * N_ * H_ * DH_) return;
  int d = (int)(idx % DH_);
  int h = (int)((idx / DH_) % H_);
  int n = (int)((idx / (DH_ * H_)) % N_);
  int b = (int)(idx / ((long)DH_ * H_ * N_));
  out[((long)(b * H_ + h) * N_ + n) * DH_ + d] = (_Float16)in[idx];
}

__global__ void __launch_bounds__(256) bhnd2bnhd_k(const float* __restrict__ in,
                                                   _Float16* __restrict__ out) {
  long idx = (long)blockIdx.x * 256 + threadIdx.x;
  if (idx >= (long)B_ * H_ * N_ * DH_) return;
  int d = (int)(idx % DH_);
  int n = (int)((idx / DH_) % N_);
  int h = (int)((idx / (DH_ * N_)) % H_);
  int b = (int)(idx / ((long)DH_ * N_ * H_));
  out[((long)(b * N_ + n) * H_ + h) * DH_ + d] = (_Float16)in[idx];
}

// rows hold anorm-scaled data, so diag = 0.5 * ||row_scaled||^2
// (since 0.5 * DH^-0.5 * ||row||^2 == 0.5 * ||anorm*row||^2)
__global__ void __launch_bounds__(256) diag_k(const _Float16* __restrict__ x,
                                              float* __restrict__ diag, long rows) {
  long row = (long)blockIdx.x * 8 + (threadIdx.x >> 5);
  int lane = threadIdx.x & 31;
  if (row >= rows) return;
  const _Float16* xr = x + row * DH_;
  float s = 0.f;
  for (int i = lane; i < DH_; i += 32) { float t = (float)xr[i]; s += t * t; }
  s = wave_sum(s);
  if (lane == 0) diag[row] = s * 0.5f;
}

// qf: per-row max, then ratio*(exp(d-diag-m)+eps); update fp32 in place and
// write padded f16 copy (row stride MP_).
__global__ void __launch_bounds__(256) qf_k(float* __restrict__ dash,
                                            _Float16* __restrict__ qf16,
                                            const float* __restrict__ diag, long rows) {
  long row = (long)blockIdx.x * 8 + (threadIdx.x >> 5);
  int lane = threadIdx.x & 31;
  if (row >= rows) return;
  float* dr = dash + row * M_;
  float m = -3.0e38f;
  for (int i = lane; i < M_; i += 32) m = fmaxf(m, dr[i]);
  m = wave_max(m);
  float dg = diag[row];
  float ratio = rsqrtf((float)M_);
  _Float16* qo = qf16 + row * MP_;
  for (int i = lane; i < M_; i += 32) {
    float val = ratio * (expf(dr[i] - dg - m) + 1e-4f);
    dr[i] = val;
    qo[i] = (_Float16)val;
  }
}

__global__ void __launch_bounds__(256) gmax1_k(const float* __restrict__ x, long n,
                                               float* __restrict__ partial) {
  __shared__ float sm[256];
  float m = -3.0e38f;
  for (long i = (long)blockIdx.x * 256 + threadIdx.x; i < n; i += (long)gridDim.x * 256)
    m = fmaxf(m, x[i]);
  sm[threadIdx.x] = m;
  __syncthreads();
  for (int s = 128; s; s >>= 1) {
    if (threadIdx.x < s) sm[threadIdx.x] = fmaxf(sm[threadIdx.x], sm[threadIdx.x + s]);
    __syncthreads();
  }
  if (threadIdx.x == 0) partial[blockIdx.x] = sm[0];
}

__global__ void __launch_bounds__(256) gmax2_k(const float* __restrict__ partial, int n,
                                               float* __restrict__ out) {
  __shared__ float sm[256];
  float m = -3.0e38f;
  for (int i = threadIdx.x; i < n; i += 256) m = fmaxf(m, partial[i]);
  sm[threadIdx.x] = m;
  __syncthreads();
  for (int s = 128; s; s >>= 1) {
    if (threadIdx.x < s) sm[threadIdx.x] = fmaxf(sm[threadIdx.x], sm[threadIdx.x + s]);
    __syncthreads();
  }
  if (threadIdx.x == 0) out[0] = sm[0];
}

// kf: global max; write transposed f16 kfT[bh][m][n] (row stride N_)
__global__ void __launch_bounds__(256) kf_k(const float* __restrict__ dash,
                                            _Float16* __restrict__ kfT,
                                            const float* __restrict__ diag,
                                            const float* __restrict__ gmax, long rows) {
  long row = (long)blockIdx.x * 8 + (threadIdx.x >> 5);   // row = bh*N + n
  int lane = threadIdx.x & 31;
  if (row >= rows) return;
  const float* dr = dash + row * M_;
  long bh = row / N_;
  long n  = row % N_;
  float m = gmax[0];
  float dg = diag[row];
  float ratio = rsqrtf((float)M_);
  for (int i = lane; i < M_; i += 32) {
    float val = ratio * (expf(dr[i] - dg - m) + 1e-4f);
    kfT[(bh * M_ + i) * N_ + n] = (_Float16)val;
  }
}

// ksum[bh][m] = sum_n kfT[bh][m][n]; one wave per (bh,m) row, contiguous reads
__global__ void __launch_bounds__(256) ksum_k(const _Float16* __restrict__ kfT,
                                              float* __restrict__ ksum) {
  long row = (long)blockIdx.x * 8 + (threadIdx.x >> 5);   // bh*M + m
  int lane = threadIdx.x & 31;
  if (row >= (long)B_ * H_ * M_) return;
  const _Float16* p = kfT + row * (long)N_;
  float s = 0.f;
  for (int i = lane; i < N_; i += 32) s += (float)p[i];
  s = wave_sum(s);
  if (lane == 0) ksum[row] = s;
}

// dinv[row] = 1 / dot(qf_row, ksum[bh]); one wave per row
__global__ void __launch_bounds__(256) dinv_k(const float* __restrict__ qf,
                                              const float* __restrict__ ksum,
                                              float* __restrict__ dinv) {
  long row = (long)blockIdx.x * 8 + (threadIdx.x >> 5);
  int lane = threadIdx.x & 31;
  if (row >= (long)B_ * H_ * N_) return;
  long bh = row / N_;
  const float* qr = qf + row * M_;
  const float* ks = ksum + bh * M_;
  float s = 0.f;
  for (int i = lane; i < M_; i += 32) s += qr[i] * ks[i];
  s = wave_sum(s);
  if (lane == 0) dinv[row] = 1.f / s;
}

// ---------------------------------------------------------------------------
// Host launcher
// ---------------------------------------------------------------------------
static inline void launch_gemm(hipStream_t s,
    const _Float16* A, const _Float16* Bm, float* C, _Float16* C16,
    int Md, int Nd, int Kd,
    long ars, long abs_, long brs, long bbs,
    long cbs, long c16rs, long c16bs, int batch,
    const float* bias, const float* rscale, long rsbs,
    const float* resid, long resbs, int gelu) {
  dim3 g((Nd + TN - 1) / TN, (Md + TM - 1) / TM, batch);
  gemm_wmma_k<<<g, 256, 0, s>>>(A, Bm, C, C16, Md, Nd, Kd, ars, abs_, brs, bbs,
                                cbs, c16rs, c16bs, bias, rscale, rsbs,
                                resid, resbs, gelu);
}

extern "C" void kernel_launch(void* const* d_in, const int* in_sizes, int n_in,
                              void* d_out, int out_size, void* d_ws, size_t ws_size,
                              hipStream_t stream) {
  (void)in_sizes; (void)n_in; (void)out_size; (void)ws_size;
  const float* x_in  = (const float*)d_in[0];
  const float* Wq    = (const float*)d_in[1];
  const float* Wk    = (const float*)d_in[2];
  const float* Wv    = (const float*)d_in[3];
  const float* Wo    = (const float*)d_in[4];
  const float* proj  = (const float*)d_in[5];
  const float* ln1g  = (const float*)d_in[6];
  const float* ln1b  = (const float*)d_in[7];
  const float* W1    = (const float*)d_in[8];
  const float* b1    = (const float*)d_in[9];
  const float* W2    = (const float*)d_in[10];
  const float* b2    = (const float*)d_in[11];
  const float* ln2g  = (const float*)d_in[12];
  const float* ln2b  = (const float*)d_in[13];
  const float* lnfg  = (const float*)d_in[14];
  const float* lnfb  = (const float*)d_in[15];
  float* out = (float*)d_out;

  const long BN   = (long)B_ * N_;                 // 8192
  const long BHN  = (long)B_ * H_ * N_;            // 65536
  const size_t S  = (size_t)B_ * N_ * D_;          // 4,194,304 elems

  // workspace carving (unit: halfs; chunks rounded up so staging over-reads
  // stay in-bounds and fp32 chunks stay 4B-aligned)
  size_t offH = 0;
  _Float16* base16 = (_Float16*)d_ws;
  auto alloc16 = [&](size_t nh) {
    _Float16* p = base16 + offH;
    offH += (nh + 1023) & ~(size_t)1023;
    return p;
  };
  auto allocF = [&](size_t nf) {
    float* p = (float*)(base16 + offH);
    offH += (2 * nf + 1023) & ~(size_t)1023;
    return p;
  };

  float*    xw    = allocF(S);
  float*    q     = allocF(S);
  float*    k     = allocF(S);
  float*    v     = allocF(S);
  float*    qdash = allocF((size_t)BHN * M_);
  float*    kdash = allocF((size_t)BHN * M_);
  float*    diagq = allocF(BHN);
  float*    diagk = allocF(BHN);
  float*    dinv  = allocF(BHN);
  float*    ksum  = allocF((size_t)B_ * H_ * M_);
  float*    part  = allocF(1024);
  float*    gmax  = allocF(1);

  _Float16* h16   = alloc16(S);
  _Float16* qt16  = alloc16(S);
  _Float16* kt16  = alloc16(S);
  _Float16* vt16  = alloc16(S);
  _Float16* qf16  = alloc16((size_t)BHN * MP_);
  _Float16* kfT16 = alloc16((size_t)B_ * H_ * M_ * N_);
  _Float16* fc116 = alloc16((size_t)B_ * N_ * FF_);
  _Float16* ctx16 = alloc16((size_t)B_ * H_ * M_ * DH_);
  _Float16* o16   = alloc16(S);
  _Float16* wq16  = alloc16((size_t)D_ * H_ * DH_);
  _Float16* wk16  = alloc16((size_t)D_ * H_ * DH_);
  _Float16* wv16  = alloc16((size_t)D_ * H_ * DH_);
  _Float16* wo16  = alloc16((size_t)H_ * DH_ * D_);
  _Float16* w116  = alloc16((size_t)D_ * FF_);
  _Float16* w216  = alloc16((size_t)FF_ * D_);
  _Float16* pjT16 = alloc16((size_t)DH_ * MP_);

  float* attn = q;   // reuse: q fp32 dead after rotary transpose

  const long pairs = (long)B_ * N_ * H_ * (DH_ / 2);
  const long elems = (long)B_ * N_ * H_ * DH_;
  const long WSZ = (long)D_ * H_ * DH_;        // 262144
  const long FSZ = (long)D_ * FF_;             // 1048576

  posadd_k<<<(int)((S + 255) / 256), 256, 0, stream>>>(x_in, xw);

  for (int i = 0; i < L_; ++i) {
    // convert this layer's weights to f16 (L2-resident afterwards)
    cvt_k<<<(int)((WSZ + 255) / 256), 256, 0, stream>>>(Wq + (size_t)i * WSZ, wq16, WSZ);
    cvt_k<<<(int)((WSZ + 255) / 256), 256, 0, stream>>>(Wk + (size_t)i * WSZ, wk16, WSZ);
    cvt_k<<<(int)((WSZ + 255) / 256), 256, 0, stream>>>(Wv + (size_t)i * WSZ, wv16, WSZ);
    cvt_k<<<(int)((WSZ + 255) / 256), 256, 0, stream>>>(Wo + (size_t)i * WSZ, wo16, WSZ);
    cvt_k<<<(int)((FSZ + 255) / 256), 256, 0, stream>>>(W1 + (size_t)i * FSZ, w116, FSZ);
    cvt_k<<<(int)((FSZ + 255) / 256), 256, 0, stream>>>(W2 + (size_t)i * FSZ, w216, FSZ);
    projT_k<<<(DH_ * M_ + 255) / 256, 256, 0, stream>>>(proj + (size_t)i * M_ * DH_, pjT16);

    // h16 = LN1(x)
    layernorm_k<<<(int)((BN + 7) / 8), 256, 0, stream>>>(
        xw, ln1g + (size_t)i * D_, ln1b + (size_t)i * D_, nullptr, h16, (int)BN, D_);

    // q,k,v = h @ W*  (fp32 out, consumed by rotary/transpose)
    launch_gemm(stream, h16, wq16, q, nullptr, (int)BN, D_, D_,
                D_, 0, D_, 0, 0, 0, 0, 1, nullptr, nullptr, 0, nullptr, 0, 0);
    launch_gemm(stream, h16, wk16, k, nullptr, (int)BN, D_, D_,
                D_, 0, D_, 0, 0, 0, 0, 1, nullptr, nullptr, 0, nullptr, 0, 0);
    launch_gemm(stream, h16, wv16, v, nullptr, (int)BN, D_, D_,
                D_, 0, D_, 0, 0, 0, 0, 1, nullptr, nullptr, 0, nullptr, 0, 0);

    // rotary (pre-scaled by anorm) + transpose to (b,h,n,d); v transpose only
    rotary_tr_k<<<(int)((pairs + 255) / 256), 256, 0, stream>>>(q, qt16);
    rotary_tr_k<<<(int)((pairs + 255) / 256), 256, 0, stream>>>(k, kt16);
    bnhd2bhnd_k<<<(int)((elems + 255) / 256), 256, 0, stream>>>(v, vt16);

    diag_k<<<(int)((BHN + 7) / 8), 256, 0, stream>>>(qt16, diagq, BHN);
    diag_k<<<(int)((BHN + 7) / 8), 256, 0, stream>>>(kt16, diagk, BHN);

    // dash = (anorm*data) @ proj^T  (batched over b*h; scale already folded)
    launch_gemm(stream, qt16, pjT16, qdash, nullptr, N_, M_, DH_,
                DH_, (long)N_ * DH_, MP_, 0, (long)N_ * M_, 0, 0, B_ * H_,
                nullptr, nullptr, 0, nullptr, 0, 0);
    launch_gemm(stream, kt16, pjT16, kdash, nullptr, N_, M_, DH_,
                DH_, (long)N_ * DH_, MP_, 0, (long)N_ * M_, 0, 0, B_ * H_,
                nullptr, nullptr, 0, nullptr, 0, 0);

    // feature maps
    qf_k<<<(int)((BHN + 7) / 8), 256, 0, stream>>>(qdash, qf16, diagq, BHN);
    gmax1_k<<<1024, 256, 0, stream>>>(kdash, (long)BHN * M_, part);
    gmax2_k<<<1, 256, 0, stream>>>(part, 1024, gmax);
    kf_k<<<(int)((BHN + 7) / 8), 256, 0, stream>>>(kdash, kfT16, diagk, gmax, BHN);

    // linear attention
    ksum_k<<<(int)(((long)B_ * H_ * M_ + 7) / 8), 256, 0, stream>>>(kfT16, ksum);
    // context16[bh] = kf^T @ v  (266 x 64, K = 2048)
    launch_gemm(stream, kfT16, vt16, nullptr, ctx16, M_, DH_, N_,
                N_, (long)M_ * N_, DH_, (long)N_ * DH_,
                0, DH_, (long)M_ * DH_, B_ * H_,
                nullptr, nullptr, 0, nullptr, 0, 0);
    dinv_k<<<(int)((BHN + 7) / 8), 256, 0, stream>>>(qdash, ksum, dinv);
    // attn[bh] = (qf @ context) * dinv_row  (2048 x 64, K = 266)
    launch_gemm(stream, qf16, ctx16, attn, nullptr, N_, DH_, M_,
                MP_, (long)N_ * MP_, DH_, (long)M_ * DH_,
                (long)N_ * DH_, 0, 0, B_ * H_,
                nullptr, dinv, N_, nullptr, 0, 0);

    // merge heads (f16) and output projection with residual (in-place on x)
    bhnd2bnhd_k<<<(int)((elems + 255) / 256), 256, 0, stream>>>(attn, o16);
    launch_gemm(stream, o16, wo16, xw, nullptr, (int)BN, D_, H_ * DH_,
                H_ * DH_, 0, D_, 0, 0, 0, 0, 1,
                nullptr, nullptr, 0, xw, 0, 0);

    // FFN
    layernorm_k<<<(int)((BN + 7) / 8), 256, 0, stream>>>(
        xw, ln2g + (size_t)i * D_, ln2b + (size_t)i * D_, nullptr, h16, (int)BN, D_);
    launch_gemm(stream, h16, w116, nullptr, fc116, (int)BN, FF_, D_,
                D_, 0, FF_, 0, 0, FF_, 0, 1,
                b1 + (size_t)i * FF_, nullptr, 0, nullptr, 0, 1);
    launch_gemm(stream, fc116, w216, xw, nullptr, (int)BN, D_, FF_,
                FF_, 0, D_, 0, 0, 0, 0, 1,
                b2 + (size_t)i * D_, nullptr, 0, xw, 0, 0);
  }

  layernorm_k<<<(int)((BN + 7) / 8), 256, 0, stream>>>(
      xw, lnfg, lnfb, out, nullptr, (int)BN, D_);
}
